// GetCost_28698971472075
// MI455X (gfx1250) — matrix operand order repaired and checked by reading, not compiled, not run on hardware
//
#include <hip/hip_runtime.h>
#include <hip/hip_bf16.h>

// ---------------------------------------------------------------------------
// GetCost (MVS plane-sweep cost volume) for MI455X / gfx1250, wave32.
//
// Shapes (fixed by the harness): V=5, B=2, C=32, H=256, W=320, CostNum=4.
// Memory-bound gather kernel: features (105MB) are L2-resident (192MB L2).
// Strategy:
//   pass0: build per-(b,view) 3x4 homographies on device (ws).
//   pass1: transpose features NCHW f32 -> NHWC f16 (ws): a pixel's 32-ch
//          vector becomes one contiguous 64B column -> 2x b128 per WMMA
//          operand half, and halves gather bytes.
//   main : one wave owns 16 pixels (one image row segment; W%16==0 so the
//          segment never crosses a row -> Y is wave-uniform, X = X0+p).
//          A = 16 ref rows (f16 WMMA-A layout, loaded once). Per
//          (pixel,view): lanes gather the 16 corner columns (4 samples x
//          4 corners) as WMMA-B; one v_wmma_f32_16x16x32_f16 produces all
//          16 corner dot-products (row p of D == one VGPR). Bilinear+view
//          weights applied per lane post-D, corner reduction via 2
//          shfl_xor, view accumulation in f32. Channel-broadcast output
//          written coalesced via LDS.
// ---------------------------------------------------------------------------

typedef __attribute__((ext_vector_type(16))) _Float16 v16h;
typedef __attribute__((ext_vector_type(8)))  float    v8f;
typedef __attribute__((ext_vector_type(4)))  float    v4f;

namespace {
constexpr int kV = 5, kB = 2, kC = 32, kH = 256, kW = 320, kCN = 4;
constexpr int kHW = kH * kW;
constexpr int PIXB = 128;                         // pixels per block (8 waves x 16)
constexpr size_t FEATT_OFF   = 4096;              // mats live at ws[0..383]
constexpr size_t FEATT_BYTES = (size_t)kV * kB * kHW * kC * 2;  // f16 NHWC copy
}

union H16 { v16h h; v4f f[2]; };

__device__ inline v16h ldpair(const _Float16* p0, const _Float16* p1) {
  H16 u;
  u.f[0] = *(const v4f*)p0;
  u.f[1] = *(const v4f*)p1;
  return u.h;
}

// ---------------------------------------------------------------------------
// pass 0: M_i = build_proj(p_i) * inv(build_proj(p_0)) ; store rows 0..2.
// build_proj(p): rows0..2 = p[1][:3,:3] @ p[0][:3,:4], row3 = p[0][3] (=0001).
// ---------------------------------------------------------------------------
__device__ inline void build_proj4(float P[4][4], const float* src) {
#pragma unroll
  for (int r = 0; r < 3; ++r)
#pragma unroll
    for (int c = 0; c < 4; ++c)
      P[r][c] = src[16 + r*4 + 0] * src[0*4 + c]
              + src[16 + r*4 + 1] * src[1*4 + c]
              + src[16 + r*4 + 2] * src[2*4 + c];
#pragma unroll
  for (int c = 0; c < 4; ++c) P[3][c] = src[12 + c];
}

__global__ __launch_bounds__(32)
void build_mats_kernel(const float* __restrict__ proj, float* __restrict__ mats) {
  const int t = threadIdx.x;
  if (t >= kB * (kV - 1)) return;
  const int b  = t / (kV - 1);
  const int iv = t % (kV - 1) + 1;
  float P0[4][4], Pi[4][4];
  build_proj4(P0, proj + (size_t)(b * kV + 0)  * 32);
  build_proj4(Pi, proj + (size_t)(b * kV + iv) * 32);
  // invert affine P0 (row 3 == [0 0 0 1] for this construction)
  const float a00=P0[0][0],a01=P0[0][1],a02=P0[0][2];
  const float a10=P0[1][0],a11=P0[1][1],a12=P0[1][2];
  const float a20=P0[2][0],a21=P0[2][1],a22=P0[2][2];
  const float c00 = a11*a22 - a12*a21;
  const float c01 = a12*a20 - a10*a22;
  const float c02 = a10*a21 - a11*a20;
  const float rd  = 1.0f / (a00*c00 + a01*c01 + a02*c02);
  float I[4][4];
  I[0][0]=c00*rd; I[0][1]=(a02*a21-a01*a22)*rd; I[0][2]=(a01*a12-a02*a11)*rd;
  I[1][0]=c01*rd; I[1][1]=(a00*a22-a02*a20)*rd; I[1][2]=(a02*a10-a00*a12)*rd;
  I[2][0]=c02*rd; I[2][1]=(a01*a20-a00*a21)*rd; I[2][2]=(a00*a11-a01*a10)*rd;
#pragma unroll
  for (int r = 0; r < 3; ++r)
    I[r][3] = -(I[r][0]*P0[0][3] + I[r][1]*P0[1][3] + I[r][2]*P0[2][3]);
  I[3][0]=0.f; I[3][1]=0.f; I[3][2]=0.f; I[3][3]=1.f;
  float* Mo = mats + (size_t)(b * (kV - 1) + iv - 1) * 12;
#pragma unroll
  for (int r = 0; r < 3; ++r)
#pragma unroll
    for (int c = 0; c < 4; ++c)
      Mo[r*4 + c] = Pi[r][0]*I[0][c] + Pi[r][1]*I[1][c]
                  + Pi[r][2]*I[2][c] + Pi[r][3]*I[3][c];
}

// ---------------------------------------------------------------------------
// pass 1: NCHW f32 -> NHWC f16. Coalesced reads per channel, 64B store/lane.
// ---------------------------------------------------------------------------
__global__ __launch_bounds__(256)
void transpose_f16_kernel(const float* __restrict__ feats,
                          _Float16* __restrict__ featT) {
  const size_t idx = (size_t)blockIdx.x * 256 + threadIdx.x;
  const int pix = (int)(idx % kHW);
  const int vb  = (int)(idx / kHW);
  const float* src = feats + (size_t)vb * kC * kHW + pix;
  v16h lo, hi;
#pragma unroll
  for (int c = 0; c < 16; ++c) lo[c] = (_Float16)src[(size_t)c * kHW];
#pragma unroll
  for (int c = 0; c < 16; ++c) hi[c] = (_Float16)src[(size_t)(c + 16) * kHW];
  H16 ul, uh; ul.h = lo; uh.h = hi;
  v4f* dst = (v4f*)(featT + ((size_t)vb * kHW + pix) * kC);
  dst[0] = ul.f[0]; dst[1] = ul.f[1]; dst[2] = uh.f[0]; dst[3] = uh.f[1];
}

// ---------------------------------------------------------------------------
// main kernel: WMMA-based cost volume.
// ---------------------------------------------------------------------------
__global__ __launch_bounds__(256)
void getcost_wmma_kernel(const _Float16* __restrict__ featT,
                         const float* __restrict__ mats,
                         const float* __restrict__ dvals,
                         const float* __restrict__ ditv,
                         const float* __restrict__ vws,
                         float* __restrict__ out) {
  __shared__ float sims[PIXB * kCN];
  const int tid  = threadIdx.x;
  const int wave = tid >> 5;
  const int lane = tid & 31;
  const int l16  = lane & 15;
  const int half = lane >> 4;

  const int blocksPerB = kHW / PIXB;                 // 640
  const int b       = blockIdx.x / blocksPerB;
  const int pixbase = (blockIdx.x % blocksPerB) * PIXB;
  const int wpix    = pixbase + wave * 16;           // this wave's 16 pixels

  // W % 16 == 0 and wpix % 16 == 0 -> the 16 pixels share one row.
  const float Yf  = (float)(wpix / kW);
  const float X0f = (float)(wpix % kW);

  // per-pixel scalars, one pixel per lane-mod-16 (duplicated in both halves)
  const int mypix = wpix + l16;
  const float dv  = 1.0f / dvals[(size_t)b * kHW + mypix];   // inverse depth
  const float di  = ditv[(size_t)b * kHW + mypix];
  const float vwA = vws[((size_t)b * (kV - 1) + half)     * kHW + mypix]; // views 1,2
  const float vwB = vws[((size_t)b * (kV - 1) + 2 + half) * kHW + mypix]; // views 3,4
  const float wsum = 1e-5f + (vwA + __shfl_xor(vwA, 16, 32))
                           + (vwB + __shfl_xor(vwB, 16, 32));
  const float scalereg = 1.0f / (32.0f * wsum);      // fold channel mean + wsum

  // A operand: 16 ref rows, CDNA5 f16 A layout:
  //   lanes 0-15 (row=l16): K 0..7 then 16..23 ; lanes 16-31: K 8..15 then 24..31
  const _Float16* refcol = featT + ((size_t)b * kHW + wpix + l16) * kC; // view 0
  const v16h amat = ldpair(refcol + half * 8, refcol + half * 8 + 16);

  // lane's (sample, corner): B column q = l16 = s*4 + corner
  const float coef = (float)(l16 >> 2) * (4.0f / 3.0f) - 2.0f;  // inv-depth coeff
  const int   dx   = l16 & 1;
  const int   dy   = (l16 >> 1) & 1;
  const int   chh  = half * 16;                      // channel half offset (halfs)

  float acc[16];
#pragma unroll
  for (int p = 0; p < 16; ++p) acc[p] = 0.0f;

  for (int iv = 1; iv < kV; ++iv) {
    const float* M = mats + ((size_t)b * (kV - 1) + (iv - 1)) * 12;
    const float m00=M[0], m01=M[1], m02=M[2],  m03=M[3];
    const float m10=M[4], m11=M[5], m12=M[6],  m13=M[7];
    const float m20=M[8], m21=M[9], m22=M[10], m23=M[11];
    // row-uniform projection bases: rx(p) = rxb + p*m00, etc.
    const float rxb = m00 * X0f + m01 * Yf + m02;
    const float ryb = m10 * X0f + m11 * Yf + m12;
    const float rzb = m20 * X0f + m21 * Yf + m22;
    const float vwsel = (iv <= 2) ? vwA : vwB;
    const int   vwoff = ((iv - 1) & 1) << 4;
    const _Float16* vbase = featT + (size_t)(iv * kB + b) * kHW * kC + chh;

#pragma unroll
    for (int p = 0; p < 16; ++p) {
      const float invd = __shfl(dv, p, 32);
      const float itv  = __shfl(di, p, 32);
      const float d    = 1.0f / fmaf(coef, itv, invd);
      const float pf   = (float)p;                 // compile-time constant
      const float pxc  = fmaf(pf, m00, rxb) * d + m03;
      const float pyc  = fmaf(pf, m10, ryb) * d + m13;
      const float pzc  = fmaf(pf, m20, rzb) * d + m23;
      const float iz = 1.0f / pzc;
      const float gx = pxc * iz, gy = pyc * iz;
      const float fx0 = floorf(gx), fy0 = floorf(gy);
      const float wx1 = gx - fx0, wy1 = gy - fy0;
      const int xi = (int)fx0 + dx;
      const int yi = (int)fy0 + dy;
      const bool valid = (xi >= 0) && (xi < kW) && (yi >= 0) && (yi < kH);
      float wgt = (dx ? wx1 : 1.0f - wx1) * (dy ? wy1 : 1.0f - wy1);
      wgt = valid ? wgt : 0.0f;
      const int xc = xi < 0 ? 0 : (xi > kW - 1 ? kW - 1 : xi);
      const int yc = yi < 0 ? 0 : (yi > kH - 1 ? kH - 1 : yi);
      // B column q: corner feature vector; lane half h holds channels 8h..8h+7 then 16+8h..16+8h+7
      const _Float16* bp = vbase + ((size_t)yc * kW + xc) * kC;
      const v16h bmat = ldpair(bp - chh + half * 8, bp - chh + half * 8 + 16);
      v8f cz = {0.f, 0.f, 0.f, 0.f, 0.f, 0.f, 0.f, 0.f};
      v8f dres = __builtin_amdgcn_wmma_f32_16x16x32_f16(
          false, amat, false, bmat, (short)0, cz, false, false);
      // row p of D lives in VGPR (p&7): lanes 0-15 for p<8, lanes 16-31 for p>=8
      const float vw = __shfl(vwsel, p + vwoff, 32);
      acc[p] = fmaf(dres[p & 7], wgt * vw, acc[p]);
    }
  }

  // reduce the 4 corners per sample, scale, stage in LDS
#pragma unroll
  for (int p = 0; p < 16; ++p) {
    float r = acc[p];
    r += __shfl_xor(r, 1, 32);
    r += __shfl_xor(r, 2, 32);
    r *= __shfl(scalereg, p, 32);
    if (((lane >> 4) == (p >> 3)) && ((lane & 3) == 0))
      sims[(wave * 16 + p) * kCN + (l16 >> 2)] = r;
  }
  __syncthreads();

  // channel-broadcast write, coalesced 512B runs: out[b, c*CN+s, pix]
  const int TOT = kC * kCN * PIXB;   // 16384
  for (int e = tid; e < TOT; e += 256) {
    const int p = e & (PIXB - 1);
    const int s = (e >> 7) & 3;
    const int c = e >> 9;
    out[(((size_t)(b * kC + c) * kCN + s) * kHW) + pixbase + p] = sims[p * kCN + s];
  }
}

// ---------------------------------------------------------------------------
// fallback (ws too small for f16 copy): plain f32 gather kernel, NCHW reads.
// ---------------------------------------------------------------------------
__global__ __launch_bounds__(256)
void getcost_fallback_kernel(const float* __restrict__ feats,
                             const float* __restrict__ mats,
                             const float* __restrict__ dvals,
                             const float* __restrict__ ditv,
                             const float* __restrict__ vws,
                             float* __restrict__ out) {
  const size_t idx = (size_t)blockIdx.x * 256 + threadIdx.x;
  const int pix  = (int)(idx % kHW);
  const int rest = (int)(idx / kHW);
  const int s = rest % kCN;
  const int b = rest / kCN;
  const float X = (float)(pix % kW), Y = (float)(pix / kW);
  const float invd = 1.0f / dvals[(size_t)b * kHW + pix];
  const float itv  = ditv[(size_t)b * kHW + pix];
  const float d = 1.0f / (invd + ((float)s * (4.0f / 3.0f) - 2.0f) * itv);
  const float* refb = feats + (size_t)b * kC * kHW + pix;   // view 0
  float vol = 0.0f, wsum = 1e-5f;
  for (int iv = 1; iv < kV; ++iv) {
    const float* M = mats + ((size_t)b * (kV - 1) + (iv - 1)) * 12;
    const float pxc = (M[0]*X + M[1]*Y + M[2])  * d + M[3];
    const float pyc = (M[4]*X + M[5]*Y + M[6])  * d + M[7];
    const float pzc = (M[8]*X + M[9]*Y + M[10]) * d + M[11];
    const float iz = 1.0f / pzc;
    const float gx = pxc * iz, gy = pyc * iz;
    const float fx0 = floorf(gx), fy0 = floorf(gy);
    const float wx1 = gx - fx0, wy1 = gy - fy0;
    const int x0 = (int)fx0, y0 = (int)fy0;
    const float vx0 = (x0 >= 0     && x0 <= kW-1) ? 1.f : 0.f;
    const float vx1 = (x0+1 >= 0   && x0+1 <= kW-1) ? 1.f : 0.f;
    const float vy0 = (y0 >= 0     && y0 <= kH-1) ? 1.f : 0.f;
    const float vy1 = (y0+1 >= 0   && y0+1 <= kH-1) ? 1.f : 0.f;
    const float w00 = (1.f-wx1)*(1.f-wy1)*vx0*vy0;
    const float w01 = wx1*(1.f-wy1)*vx1*vy0;
    const float w10 = (1.f-wx1)*wy1*vx0*vy1;
    const float w11 = wx1*wy1*vx1*vy1;
    const int xc0 = x0   < 0 ? 0 : (x0   > kW-1 ? kW-1 : x0);
    const int xc1 = x0+1 < 0 ? 0 : (x0+1 > kW-1 ? kW-1 : x0+1);
    const int yc0 = y0   < 0 ? 0 : (y0   > kH-1 ? kH-1 : y0);
    const int yc1 = y0+1 < 0 ? 0 : (y0+1 > kH-1 ? kH-1 : y0+1);
    const size_t o00 = (size_t)yc0 * kW + xc0;
    const size_t o01 = (size_t)yc0 * kW + xc1;
    const size_t o10 = (size_t)yc1 * kW + xc0;
    const size_t o11 = (size_t)yc1 * kW + xc1;
    const float* fb = feats + (size_t)(iv * kB + b) * kC * kHW;
    float dot = 0.0f;
#pragma unroll 8
    for (int c = 0; c < kC; ++c) {
      const float* f = fb + (size_t)c * kHW;
      const float wp = w00*f[o00] + w01*f[o01] + w10*f[o10] + w11*f[o11];
      dot = fmaf(wp, refb[(size_t)c * kHW], dot);
    }
    const float vw = vws[((size_t)b * (kV - 1) + (iv - 1)) * kHW + pix];
    vol = fmaf(dot * (1.0f / 32.0f), vw, vol);
    wsum += vw;
  }
  const float sim = vol / wsum;
#pragma unroll
  for (int c = 0; c < kC; ++c)
    out[(((size_t)(b * kC + c) * kCN + s) * kHW) + pix] = sim;
}

// ---------------------------------------------------------------------------
extern "C" void kernel_launch(void* const* d_in, const int* in_sizes, int n_in,
                              void* d_out, int out_size, void* d_ws, size_t ws_size,
                              hipStream_t stream) {
  (void)in_sizes; (void)n_in; (void)out_size;
  const float* dvals = (const float*)d_in[0];   // depth_values  (B,1,H,W)
  const float* feats = (const float*)d_in[1];   // features      (V,B,C,H,W)
  const float* proj  = (const float*)d_in[2];   // proj_matrices (B,V,2,4,4)
  const float* ditv  = (const float*)d_in[3];   // depth_interval(B,1,H,W)
  const float* vws   = (const float*)d_in[7];   // view_weights  (B,V-1,H,W)
  float* out  = (float*)d_out;
  float* mats = (float*)d_ws;                   // 8 x 12 floats at ws[0]

  hipLaunchKernelGGL(build_mats_kernel, dim3(1), dim3(32), 0, stream, proj, mats);

  if (ws_size >= FEATT_OFF + FEATT_BYTES) {
    _Float16* featT = (_Float16*)((char*)d_ws + FEATT_OFF);
    const int tgrid = (kV * kB * kHW) / 256;     // 3200 blocks
    hipLaunchKernelGGL(transpose_f16_kernel, dim3(tgrid), dim3(256), 0, stream,
                       feats, featT);
    const int mgrid = kB * (kHW / PIXB);         // 1280 blocks
    hipLaunchKernelGGL(getcost_wmma_kernel, dim3(mgrid), dim3(256), 0, stream,
                       featT, mats, dvals, ditv, vws, out);
  } else {
    const int fgrid = (kB * kCN * kHW) / 256;    // 2560 blocks
    hipLaunchKernelGGL(getcost_fallback_kernel, dim3(fgrid), dim3(256), 0, stream,
                       feats, mats, dvals, ditv, vws, out);
  }
}
